// Transformer_24550033063991
// MI455X (gfx1250) — matrix operand order repaired and checked by reading, not compiled, not run on hardware
//
#include <hip/hip_runtime.h>
#include <math.h>

// Problem constants (match reference)
constexpr int Bn = 4, Sn = 512, Vn = 32000, Dn = 512, Hn = 8, DKn = 64, DVn = 64;
constexpr int LEn = 2, LDn = 2;
constexpr int BSn = Bn * Sn;            // 2048 token rows

typedef __attribute__((ext_vector_type(16))) __bf16 v16bf;
typedef __attribute__((ext_vector_type(8)))  __bf16 v8bf;
typedef __attribute__((ext_vector_type(2)))  __bf16 v2bf;
typedef __attribute__((ext_vector_type(8)))  float  v8f;

// pack two f32 into one dword of two bf16 (native cvt; RNE) -> v_cvt_pk_bf16_f32
static __device__ __forceinline__ unsigned pack2bf(float a, float b) {
  v2bf v = {(__bf16)a, (__bf16)b};
  return __builtin_bit_cast(unsigned, v);
}

// ---------------------------------------------------------------------------
// LDS-staged batched GEMM: C[m,n] = alpha * sum_c A[m,c] * B(c,n)
//   A row-major (lda).  B element (c,n):
//     BMODE==0 (row-contig):  Bm[c*ldb + n]       (projections, attn@V, Wo)
//     BMODE==1 (col-contig):  Bm[n*ldb + c]       (Q@K^T, logits vs emb^T)
//   batch z: b1=z/zdiv, b2=z%zdiv with per-index strides.
// Block = 128 threads = 4 waves; C tile 64x64; K-step 32.
// Each wave computes a 16x64 strip: 1 A-frag reused across 4 WMMAs per step.
// Fragments are built from bf16 tiles staged in LDS (padded rows, b128 access).
// ---------------------------------------------------------------------------
constexpr int LDS_PITCH = 40;   // ushorts per row (32 + 8 pad) -> 80B rows

template <int BMODE>
__global__ void __launch_bounds__(128) gemm_bf16_wmma(
    const float* __restrict__ A, long long lda, long long sA1, long long sA2,
    const float* __restrict__ Bm, long long ldb, long long sB1, long long sB2,
    float* __restrict__ C, long long ldc, long long sC1, long long sC2,
    int K, int zdiv, float alpha)
{
  __shared__ __align__(16) unsigned short As[64 * LDS_PITCH];
  __shared__ __align__(16) unsigned short Bs[64 * LDS_PITCH];

  const int z  = blockIdx.z;
  const int b1 = z / zdiv, b2 = z % zdiv;
  A  += (long long)b1 * sA1 + (long long)b2 * sA2;
  Bm += (long long)b1 * sB1 + (long long)b2 * sB2;
  C  += (long long)b1 * sC1 + (long long)b2 * sC2;

  const int m0 = blockIdx.y * 64;
  const int n0 = blockIdx.x * 64;

  const int tid    = threadIdx.x;
  const int w      = tid >> 5;        // wave 0..3 -> rows w*16..w*16+15
  const int l      = tid & 31;
  const int lane16 = l & 15;
  const int hi     = l >> 4;          // lane half

  v8f acc[4];
#pragma unroll
  for (int j = 0; j < 4; ++j) acc[j] = v8f{0.f,0.f,0.f,0.f,0.f,0.f,0.f,0.f};

  // staging coordinates (A and B col-contig path)
  const int sr = tid >> 1;            // 0..63 (row for A / col for B-colmajor)
  const int sh = tid & 1;             // 16-float half
  // staging coordinates (B row-contig transpose path): 2 c-rows x 8 n-cols
  const int tc = (tid >> 3) * 2;      // c base: 0,2,..,30
  const int tn = (tid & 7) * 8;       // n base: 0,8,..,56

  for (int kc = 0; kc < K; kc += 32) {
    // ---- stage A: 64x32 f32 -> bf16 LDS ----
    {
      const float4* p = reinterpret_cast<const float4*>(
          A + (long long)(m0 + sr) * lda + kc + sh * 16);
      float4 f0 = p[0], f1 = p[1], f2 = p[2], f3 = p[3];
      if (kc + 32 < K) __builtin_prefetch(((const float*)p) + 32, 0, 1);
      uint4* dst = reinterpret_cast<uint4*>(&As[sr * LDS_PITCH + sh * 16]);
      dst[0] = uint4{pack2bf(f0.x, f0.y), pack2bf(f0.z, f0.w),
                     pack2bf(f1.x, f1.y), pack2bf(f1.z, f1.w)};
      dst[1] = uint4{pack2bf(f2.x, f2.y), pack2bf(f2.z, f2.w),
                     pack2bf(f3.x, f3.y), pack2bf(f3.z, f3.w)};
    }
    // ---- stage B: -> Bs[n][c] bf16 LDS (n-major) ----
    if (BMODE == 1) {
      // col-contig: Bm[n*ldb + c] ; contiguous in c, same pattern as A
      const float4* p = reinterpret_cast<const float4*>(
          Bm + (long long)(n0 + sr) * ldb + kc + sh * 16);
      float4 f0 = p[0], f1 = p[1], f2 = p[2], f3 = p[3];
      if (kc + 32 < K) __builtin_prefetch(((const float*)p) + 32, 0, 1);
      uint4* dst = reinterpret_cast<uint4*>(&Bs[sr * LDS_PITCH + sh * 16]);
      dst[0] = uint4{pack2bf(f0.x, f0.y), pack2bf(f0.z, f0.w),
                     pack2bf(f1.x, f1.y), pack2bf(f1.z, f1.w)};
      dst[1] = uint4{pack2bf(f2.x, f2.y), pack2bf(f2.z, f2.w),
                     pack2bf(f3.x, f3.y), pack2bf(f3.z, f3.w)};
    } else {
      // row-contig: Bm[c*ldb + n]; each thread transposes 2 c-rows x 8 n-cols,
      // packing (c,c+1) pairs into dword LDS stores along the contiguous c axis.
      const float* r0 = Bm + (long long)(kc + tc) * ldb + n0 + tn;
      const float* r1 = r0 + ldb;
      const float4* p0 = reinterpret_cast<const float4*>(r0);
      const float4* p1 = reinterpret_cast<const float4*>(r1);
      float4 a0 = p0[0], a1 = p0[1];
      float4 b0 = p1[0], b1 = p1[1];
      if (kc + 32 < K) __builtin_prefetch(r0 + 32 * ldb, 0, 1);
      const float fa[8] = {a0.x,a0.y,a0.z,a0.w, a1.x,a1.y,a1.z,a1.w};
      const float fb[8] = {b0.x,b0.y,b0.z,b0.w, b1.x,b1.y,b1.z,b1.w};
#pragma unroll
      for (int j = 0; j < 8; ++j)
        *reinterpret_cast<unsigned*>(&Bs[(tn + j) * LDS_PITCH + tc]) =
            pack2bf(fa[j], fb[j]);
    }
    __syncthreads();

    // ---- compute: 1 A-frag, 4 B-frags, 4 WMMAs per wave ----
    {
      const int ar = w * 16 + lane16;
      const v8bf a0 = *reinterpret_cast<const v8bf*>(&As[ar * LDS_PITCH + 16 * hi]);
      const v8bf a1 = *reinterpret_cast<const v8bf*>(&As[ar * LDS_PITCH + 16 * hi + 8]);
      const v16bf afrag = __builtin_shufflevector(a0, a1,
          0,1,2,3,4,5,6,7,8,9,10,11,12,13,14,15);
#pragma unroll
      for (int j = 0; j < 4; ++j) {
        const int br = j * 16 + lane16;
        const v8bf b0 = *reinterpret_cast<const v8bf*>(&Bs[br * LDS_PITCH + 16 * hi]);
        const v8bf b1 = *reinterpret_cast<const v8bf*>(&Bs[br * LDS_PITCH + 16 * hi + 8]);
        const v16bf bfrag = __builtin_shufflevector(b0, b1,
            0,1,2,3,4,5,6,7,8,9,10,11,12,13,14,15);
        acc[j] = __builtin_amdgcn_wmma_f32_16x16x32_bf16(
            false, afrag, false, bfrag, (short)0, acc[j], false, false);
      }
    }
    __syncthreads();
  }

  // ---- epilogue: D layout VGPR i -> row 8*hi+i, col lane16 ----
#pragma unroll
  for (int j = 0; j < 4; ++j) {
    float* Cp = C + (long long)(m0 + w * 16 + 8 * hi) * ldc + n0 + j * 16 + lane16;
#pragma unroll
    for (int i = 0; i < 8; ++i) Cp[(long long)i * ldc] = acc[j][i] * alpha;
  }
}

// ---------------------------------------------------------------------------
// Embedding lookup * sqrt(D) + positional encoding
// ---------------------------------------------------------------------------
__global__ void __launch_bounds__(256) embed_pe_kernel(
    const int* __restrict__ idx, const float* __restrict__ emb,
    float* __restrict__ out, int total)
{
  int i = blockIdx.x * 256 + threadIdx.x;
  if (i >= total) return;
  int d  = i % Dn;
  int bs = i / Dn;
  int s  = bs % Sn;
  int tok = idx[bs];
  float x = emb[(long long)tok * Dn + d] * 22.62741699796952f; // sqrt(512)
  float num = (d & 1) ? cosf((float)s) : sinf((float)s);
  float den = powf(10000.0f, 2.0f * (float)d / (float)Dn);
  out[i] = x + num / den;
}

// ---------------------------------------------------------------------------
// Row softmax over S=512 scores (1/sqrt(dk) folded into GEMM alpha);
// causal adds -1e15 above the diagonal before softmax.
// ---------------------------------------------------------------------------
__global__ void __launch_bounds__(256) attn_softmax_kernel(float* __restrict__ scores, int causal)
{
  __shared__ float red[256];
  const int row = blockIdx.x;        // b*H*S + h*S + q
  const int q   = row % Sn;
  const int t   = threadIdx.x;
  float* p = scores + (long long)row * Sn;

  float v0 = p[t], v1 = p[t + 256];
  if (causal) {
    if (t > q)        v0 += -1e15f;
    if (t + 256 > q)  v1 += -1e15f;
  }
  red[t] = fmaxf(v0, v1); __syncthreads();
  for (int off = 128; off > 0; off >>= 1) {
    if (t < off) red[t] = fmaxf(red[t], red[t + off]);
    __syncthreads();
  }
  float m = red[0]; __syncthreads();

  float e0 = __expf(v0 - m), e1 = __expf(v1 - m);
  red[t] = e0 + e1; __syncthreads();
  for (int off = 128; off > 0; off >>= 1) {
    if (t < off) red[t] += red[t + off];
    __syncthreads();
  }
  float inv = 1.0f / red[0];
  p[t] = e0 * inv; p[t + 256] = e1 * inv;
}

// ---------------------------------------------------------------------------
// x = LayerNorm(x + r): (x-mean)/(std+eps)*g + b  (population std, matches ref)
// ---------------------------------------------------------------------------
__global__ void __launch_bounds__(256) add_ln_kernel(
    float* __restrict__ x, const float* __restrict__ r,
    const float* __restrict__ g, const float* __restrict__ b)
{
  __shared__ float red[256];
  const int row = blockIdx.x;
  const int t   = threadIdx.x;
  float* px = x + (long long)row * Dn;
  const float* pr = r + (long long)row * Dn;

  float v0 = px[t] + pr[t];
  float v1 = px[t + 256] + pr[t + 256];

  red[t] = v0 + v1; __syncthreads();
  for (int off = 128; off > 0; off >>= 1) {
    if (t < off) red[t] += red[t + off];
    __syncthreads();
  }
  float mean = red[0] * (1.0f / Dn); __syncthreads();

  float d0 = v0 - mean, d1 = v1 - mean;
  red[t] = d0 * d0 + d1 * d1; __syncthreads();
  for (int off = 128; off > 0; off >>= 1) {
    if (t < off) red[t] += red[t + off];
    __syncthreads();
  }
  float denom = sqrtf(red[0] * (1.0f / Dn)) + 1e-6f;
  float inv = 1.0f / denom;
  px[t]       = g[t]       * d0 * inv + b[t];
  px[t + 256] = g[t + 256] * d1 * inv + b[t + 256];
}

// ---------------------------------------------------------------------------
// In-place softmax over the vocab dimension (V=32000), one block per token.
// ---------------------------------------------------------------------------
__global__ void __launch_bounds__(256) vocab_softmax_kernel(float* __restrict__ out)
{
  __shared__ float red[256];
  const int row = blockIdx.x;
  const int t   = threadIdx.x;
  float* p = out + (long long)row * Vn;

  float m = -3.0e38f;
  for (int i = t; i < Vn; i += 256) m = fmaxf(m, p[i]);
  red[t] = m; __syncthreads();
  for (int off = 128; off > 0; off >>= 1) {
    if (t < off) red[t] = fmaxf(red[t], red[t + off]);
    __syncthreads();
  }
  m = red[0]; __syncthreads();

  float s = 0.f;
  for (int i = t; i < Vn; i += 256) {
    float e = __expf(p[i] - m);
    p[i] = e;
    s += e;
  }
  red[t] = s; __syncthreads();
  for (int off = 128; off > 0; off >>= 1) {
    if (t < off) red[t] += red[t + off];
    __syncthreads();
  }
  float inv = 1.0f / red[0];
  for (int i = t; i < Vn; i += 256) p[i] *= inv;
}

// ---------------------------------------------------------------------------
extern "C" void kernel_launch(void* const* d_in, const int* in_sizes, int n_in,
                              void* d_out, int out_size, void* d_ws, size_t ws_size,
                              hipStream_t stream) {
  (void)in_sizes; (void)n_in; (void)out_size; (void)ws_size;

  const int*   src    = (const int*)  d_in[0];
  const int*   tgt    = (const int*)  d_in[1];
  const float* emb    = (const float*)d_in[2];
  const float* enc_Wq = (const float*)d_in[3];
  const float* enc_Wk = (const float*)d_in[4];
  const float* enc_Wv = (const float*)d_in[5];
  const float* enc_Wo = (const float*)d_in[6];
  const float* enc_g  = (const float*)d_in[7];
  const float* enc_b  = (const float*)d_in[8];
  const float* dWq1   = (const float*)d_in[9];
  const float* dWk1   = (const float*)d_in[10];
  const float* dWv1   = (const float*)d_in[11];
  const float* dWo1   = (const float*)d_in[12];
  const float* dg1    = (const float*)d_in[13];
  const float* db1    = (const float*)d_in[14];
  const float* dWq2   = (const float*)d_in[15];
  const float* dWk2   = (const float*)d_in[16];
  const float* dWv2   = (const float*)d_in[17];
  const float* dWo2   = (const float*)d_in[18];
  const float* dg2    = (const float*)d_in[19];
  const float* db2    = (const float*)d_in[20];
  float* out = (float*)d_out;

  // Workspace carve-up (floats)
  float* ws      = (float*)d_ws;
  float* xbuf    = ws;                              // [B,S,D]     1M
  float* ybuf    = xbuf   + (long long)BSn * Dn;    // [B,S,D]     1M
  float* tmp     = ybuf   + (long long)BSn * Dn;    // [B,S,D]     1M
  float* qp      = tmp    + (long long)BSn * Dn;    // [B,H,S,DK]  1M
  float* kp      = qp     + (long long)Bn * Hn * Sn * DKn;
  float* vp      = kp     + (long long)Bn * Hn * Sn * DKn;
  float* attnout = vp     + (long long)Bn * Hn * Sn * DVn; // [B,S,H*DV] 1M
  float* scores  = attnout+ (long long)BSn * Hn * DVn;     // [B,H,S,S]  8M

  const long long SD   = (long long)Sn * Dn;        // per-batch activation stride
  const long long HSK  = (long long)Hn * Sn * DKn;  // per-batch head-proj stride
  const long long SK   = (long long)Sn * DKn;       // per-head  head-proj stride
  const long long HSS  = (long long)Hn * Sn * Sn;   // per-batch scores stride
  const long long SS   = (long long)Sn * Sn;        // per-head  scores stride
  const long long DDK  = (long long)Dn * DKn;       // per-head weight stride

  // bmode: 0 = B row-contig (Bm[c*ldb+n]), 1 = B col-contig (Bm[n*ldb+c])
  auto gemm = [&](int bmode,
                  const float* A, long long lda, long long sA1, long long sA2,
                  const float* Bm, long long ldb, long long sB1, long long sB2,
                  float* C, long long ldc, long long sC1, long long sC2,
                  int M, int N, int nb, int K, int zdiv, float alpha) {
    dim3 grid(N / 64, M / 64, nb);
    if (bmode == 0)
      gemm_bf16_wmma<0><<<grid, 128, 0, stream>>>(
          A, lda, sA1, sA2, Bm, ldb, sB1, sB2, C, ldc, sC1, sC2, K, zdiv, alpha);
    else
      gemm_bf16_wmma<1><<<grid, 128, 0, stream>>>(
          A, lda, sA1, sA2, Bm, ldb, sB1, sB2, C, ldc, sC1, sC2, K, zdiv, alpha);
  };

  // One multi-head attention block: dst = MHA(qsrc; kvsrc) @ Wo
  auto run_mha = [&](const float* qsrc, const float* kvsrc,
                     const float* Wq, const float* Wk, const float* Wv, const float* Wo,
                     int causal, float* dst) {
    // Q/K/V projections: per (b,h): [S,D] x [D,DK] -> [S,DK]   (B row-contig, ldb=DK)
    gemm(0, qsrc,  Dn, SD, 0,  Wq, DKn, 0, DDK,  qp, DKn, HSK, SK,
         Sn, DKn, Bn * Hn, Dn, Hn, 1.0f);
    gemm(0, kvsrc, Dn, SD, 0,  Wk, DKn, 0, DDK,  kp, DKn, HSK, SK,
         Sn, DKn, Bn * Hn, Dn, Hn, 1.0f);
    gemm(0, kvsrc, Dn, SD, 0,  Wv, DVn, 0, DDK,  vp, DVn, HSK, SK,
         Sn, DVn, Bn * Hn, Dn, Hn, 1.0f);
    // scores = (qp @ kp^T)/8 : B(c=k,n=s)=kp[s*DK+k] -> col-contig, ldb=DK
    gemm(1, qp, DKn, HSK, SK,  kp, DKn, HSK, SK,  scores, Sn, HSS, SS,
         Sn, Sn, Bn * Hn, DKn, Hn, 0.125f);
    attn_softmax_kernel<<<Bn * Hn * Sn, 256, 0, stream>>>(scores, causal);
    // attnout[b,s,h*64+v] = attn @ vp  (B row-contig, ldb=DV)
    gemm(0, scores, Sn, HSS, SS,  vp, DVn, HSK, SK,
         attnout, (long long)Hn * DVn, (long long)Sn * Hn * DVn, DVn,
         Sn, DVn, Bn * Hn, Sn, Hn, 1.0f);
    // dst = attnout @ Wo : [B*S, H*DV] x [H*DV, D]  (B row-contig, ldb=D)
    gemm(0, attnout, (long long)Hn * DVn, 0, 0,  Wo, Dn, 0, 0,  dst, Dn, 0, 0,
         BSn, Dn, 1, Hn * DVn, 1, 1.0f);
  };

  const int embBlocks = (BSn * Dn + 255) / 256;

  // ---------------- encoder ----------------
  embed_pe_kernel<<<embBlocks, 256, 0, stream>>>(src, emb, xbuf, BSn * Dn);
  for (int i = 0; i < LEn; ++i) {
    run_mha(xbuf, xbuf,
            enc_Wq + (long long)i * Hn * Dn * DKn, enc_Wk + (long long)i * Hn * Dn * DKn,
            enc_Wv + (long long)i * Hn * Dn * DVn, enc_Wo + (long long)i * Hn * DVn * Dn,
            0, tmp);
    add_ln_kernel<<<BSn, 256, 0, stream>>>(xbuf, tmp, enc_g + i * Dn, enc_b + i * Dn);
  }

  // ---------------- decoder ----------------
  embed_pe_kernel<<<embBlocks, 256, 0, stream>>>(tgt, emb, ybuf, BSn * Dn);
  for (int i = 0; i < LDn; ++i) {
    run_mha(ybuf, ybuf,
            dWq1 + (long long)i * Hn * Dn * DKn, dWk1 + (long long)i * Hn * Dn * DKn,
            dWv1 + (long long)i * Hn * Dn * DVn, dWo1 + (long long)i * Hn * DVn * Dn,
            1, tmp);
    add_ln_kernel<<<BSn, 256, 0, stream>>>(ybuf, tmp, dg1 + i * Dn, db1 + i * Dn);
    run_mha(ybuf, xbuf,
            dWq2 + (long long)i * Hn * Dn * DKn, dWk2 + (long long)i * Hn * Dn * DKn,
            dWv2 + (long long)i * Hn * Dn * DVn, dWo2 + (long long)i * Hn * DVn * Dn,
            0, tmp);
    add_ln_kernel<<<BSn, 256, 0, stream>>>(ybuf, tmp, dg2 + i * Dn, db2 + i * Dn);
  }

  // ---------------- logits + output softmax ----------------
  // out = y @ emb^T : B(c=d, n=v) = emb[v*D + d] -> col-contig, ldb=D
  gemm(1, ybuf, Dn, 0, 0,  emb, Dn, 0, 0,  out, Vn, 0, 0,
       BSn, Vn, 1, Dn, 1, 1.0f);
  vocab_softmax_kernel<<<BSn, 256, 0, stream>>>(out);
}
